// LSTM_65755949302104
// MI455X (gfx1250) — matrix operand order repaired and checked by reading, not compiled, not run on hardware
//
#include <hip/hip_runtime.h>

typedef __attribute__((ext_vector_type(16))) __bf16 bf16x16;
typedef __attribute__((ext_vector_type(8)))  float  f32x8;

static constexpr int BB = 512;   // batch
static constexpr int SS = 128;   // seq len
static constexpr int II = 256;   // input features
static constexpr int HH = 512;   // hidden

// ---------------- workspace layout (bytes) ----------------
static constexpr size_t SZ_HS0  = (size_t)BB * SS * HH * 2;     // layer0 hidden seq, bf16
static constexpr size_t SZ_WIH0 = (size_t)4 * HH * II * 2;      // 2048x256 bf16
static constexpr size_t SZ_W2   = (size_t)4 * HH * HH * 2;      // 2048x512 bf16
static constexpr size_t OFF_HS0  = 0;
static constexpr size_t OFF_WIH0 = OFF_HS0 + SZ_HS0;
static constexpr size_t OFF_WHH0 = OFF_WIH0 + SZ_WIH0;
static constexpr size_t OFF_WIH1 = OFF_WHH0 + SZ_W2;
static constexpr size_t OFF_WHH1 = OFF_WIH1 + SZ_W2;
static constexpr size_t OFF_B0   = OFF_WHH1 + SZ_W2;
static constexpr size_t OFF_B1   = OFF_B0 + (size_t)4 * HH * 4;

// ---------------- small prep kernels ----------------
__global__ void cvt_f32_bf16(const float* __restrict__ in, __bf16* __restrict__ out, int n) {
  int i = blockIdx.x * 256 + threadIdx.x;
  if (i < n) out[i] = (__bf16)in[i];
}

__global__ void bias_combine(const float* __restrict__ a, const float* __restrict__ b,
                             float* __restrict__ o, int n) {
  int i = blockIdx.x * 256 + threadIdx.x;
  if (i < n) o[i] = a[i] + b[i];
}

// ---------------- fragment loaders ----------------
// A: 16x32 bf16 tile (rows = batch), row-major in LDS, row stride = rowStride halves.
// Lanes 0-15: V0..3 hold K=k0+0..7, V4..7 hold K=k0+16..23 (M = lane).
// Lanes 16-31: same M, K ranges shifted by 8.
__device__ __forceinline__ bf16x16 load_frag_a(const __bf16* base, int rowStride,
                                               int k0, int ln, int hi) {
  union { bf16x16 v; uint4 q[2]; } u;
  const __bf16* p = base + ln * rowStride + k0 + hi * 8;
  u.q[0] = *(const uint4*)(p);
  u.q[1] = *(const uint4*)(p + 16);
  return u.v;
}

// B: 32x16 bf16 (KxN), B[k][n] = W[nbase+n][k] (W row-major, rowStride halves).
// Lanes 0-15: V0..3 hold K=k0+0..7, V4..7 hold K=k0+16..23 (N = lane); lanes 16-31: K ranges shifted by 8.
__device__ __forceinline__ bf16x16 load_frag_b(const __bf16* W, int rowStride,
                                               int nbase, int k0, int ln, int hi) {
  union { bf16x16 v; uint4 q[2]; } u;
  const __bf16* p = W + (size_t)(nbase + ln) * rowStride + k0 + hi * 8;
  u.q[0] = *(const uint4*)(p);
  u.q[1] = *(const uint4*)(p + 16);
  return u.v;
}

__device__ __forceinline__ float sigmoidf_(float x) {
  return 1.0f / (1.0f + __expf(-x));
}

// ---------------- persistent per-batch-tile LSTM layer ----------------
// Grid: 32 blocks (B/16 batch tiles) x 512 threads (16 waves).
// Wave w owns hidden units j in [w*32, w*32+32) -> 2 j-tiles x 4 gates of 16x16 f32 accums.
template <int IN_K, bool IN_F32, bool WRITE_HS, bool DO_FC>
__global__ __launch_bounds__(512) void lstm_layer(
    const float*  __restrict__ xf,     // layer0 input (f32) or null
    const __bf16* __restrict__ xb,     // layer1 input (bf16 hidden seq) or null
    const __bf16* __restrict__ Wih,    // [4H, IN_K] bf16
    const __bf16* __restrict__ Whh,    // [4H, H]    bf16
    const float*  __restrict__ bias,   // [4H] = b_ih + b_hh
    __bf16*       __restrict__ hs_out, // [B,S,H] bf16 (layer0 only)
    const float*  __restrict__ Wfc,    // [H] (layer1 only)
    const float*  __restrict__ bfc,    // [1]
    float*        __restrict__ out)    // [B] (layer1 only)
{
  __shared__ __align__(16) __bf16 lds_x[16 * IN_K];
  __shared__ __align__(16) __bf16 lds_h[16 * HH];
  __shared__ float lds_red[512];

  const int tid   = threadIdx.x;
  const int lane  = tid & 31;
  const int w     = tid >> 5;        // wave id 0..15
  const int hi    = (lane >> 4) & 1;
  const int ln    = lane & 15;
  const int btile = blockIdx.x;      // 0..31
  const int jb    = w * 32;          // this wave's hidden-unit base

  // zero initial hidden state
  for (int i = tid; i < 16 * HH; i += 512) lds_h[i] = (__bf16)0.0f;

  f32x8 acc[2][4];     // [j-tile][gate i,f,g,o]
  float cst[2][8];     // cell state, same (m,n) lane mapping as accumulators
#pragma unroll
  for (int jt = 0; jt < 2; ++jt)
#pragma unroll
    for (int r = 0; r < 8; ++r) cst[jt][r] = 0.0f;

  __syncthreads();

  for (int t = 0; t < SS; ++t) {
    // ---- stage x_t tile [16, IN_K] into LDS (bf16) ----
    {
      const int m = tid >> 5, part = tid & 31;   // 32 threads per row
      if constexpr (IN_F32) {
        const float* src = xf + ((size_t)(btile * 16 + m) * SS + t) * IN_K;
        for (int j = part; j < IN_K; j += 32) lds_x[m * IN_K + j] = (__bf16)src[j];
        if (t + 1 < SS) __builtin_prefetch(src + IN_K, 0, 3);
      } else {
        const __bf16* src = xb + ((size_t)(btile * 16 + m) * SS + t) * IN_K;
        uint4* dq = (uint4*)&lds_x[m * IN_K];
        const uint4* sq = (const uint4*)src;
#pragma unroll
        for (int q = 0; q < (IN_K / 8) / 32; ++q) dq[part + q * 32] = sq[part + q * 32];
        if (t + 1 < SS) __builtin_prefetch(src + IN_K, 0, 3);
      }
    }
    __syncthreads();

    // ---- init accumulators with combined bias ----
#pragma unroll
    for (int jt = 0; jt < 2; ++jt)
#pragma unroll
      for (int g = 0; g < 4; ++g) {
        const float bv = bias[g * HH + jb + jt * 16 + ln];
#pragma unroll
        for (int r = 0; r < 8; ++r) acc[jt][g][r] = bv;
      }

    // ---- gates += x_t @ Wih^T ----
    // Keep the k-loop rolled: prevents LICM from hoisting (and spilling) all
    // weight fragments out of the timestep loop.
#pragma unroll 1
    for (int k0 = 0; k0 < IN_K; k0 += 32) {
      const bf16x16 a = load_frag_a(lds_x, IN_K, k0, ln, hi);
      bf16x16 bfr[2][4];
#pragma unroll
      for (int jt = 0; jt < 2; ++jt)
#pragma unroll
        for (int g = 0; g < 4; ++g)
          bfr[jt][g] = load_frag_b(Wih, IN_K, g * HH + jb + jt * 16, k0, ln, hi);
#pragma unroll
      for (int jt = 0; jt < 2; ++jt)
#pragma unroll
        for (int g = 0; g < 4; ++g)
          acc[jt][g] = __builtin_amdgcn_wmma_f32_16x16x32_bf16(
              false, a, false, bfr[jt][g], (short)0, acc[jt][g], false, false);
    }

    // ---- gates += h_{t-1} @ Whh^T ----
#pragma unroll 1
    for (int k0 = 0; k0 < HH; k0 += 32) {
      const bf16x16 a = load_frag_a(lds_h, HH, k0, ln, hi);
      bf16x16 bfr[2][4];
#pragma unroll
      for (int jt = 0; jt < 2; ++jt)
#pragma unroll
        for (int g = 0; g < 4; ++g)
          bfr[jt][g] = load_frag_b(Whh, HH, g * HH + jb + jt * 16, k0, ln, hi);
#pragma unroll
      for (int jt = 0; jt < 2; ++jt)
#pragma unroll
        for (int g = 0; g < 4; ++g)
          acc[jt][g] = __builtin_amdgcn_wmma_f32_16x16x32_bf16(
              false, a, false, bfr[jt][g], (short)0, acc[jt][g], false, false);
    }
    __syncthreads();   // all lds_h reads done before overwrite

    // ---- elementwise gate math directly on accumulators ----
#pragma unroll
    for (int jt = 0; jt < 2; ++jt) {
      const int j = jb + jt * 16 + ln;
#pragma unroll
      for (int r = 0; r < 8; ++r) {
        const float iv = acc[jt][0][r];
        const float fv = acc[jt][1][r];
        const float gv = acc[jt][2][r];
        const float ov = acc[jt][3][r];
        const float cn = sigmoidf_(fv) * cst[jt][r] + sigmoidf_(iv) * tanhf(gv);
        cst[jt][r] = cn;
        const float hv = sigmoidf_(ov) * tanhf(cn);
        const int m = r + hi * 8;            // C/D layout: VGPR r -> M=r (+8 for hi lanes)
        lds_h[m * HH + j] = (__bf16)hv;
        if constexpr (WRITE_HS)
          hs_out[((size_t)(btile * 16 + m) * SS + t) * HH + j] = (__bf16)hv;
      }
    }
    __syncthreads();
  }

  // ---- fused FC (C == 1): out[b] = h_last[b,:] . Wfc + bfc ----
  if constexpr (DO_FC) {
    const int m = tid >> 5, part = tid & 31;
    float p = 0.0f;
    for (int j = part; j < HH; j += 32) p += (float)lds_h[m * HH + j] * Wfc[j];
    lds_red[tid] = p;
    __syncthreads();
    if (part == 0) {
      float s = 0.0f;
#pragma unroll
      for (int k = 0; k < 32; ++k) s += lds_red[m * 32 + k];
      out[btile * 16 + m] = s + bfc[0];
    }
  }
}

// ---------------- host launcher ----------------
extern "C" void kernel_launch(void* const* d_in, const int* in_sizes, int n_in,
                              void* d_out, int out_size, void* d_ws, size_t ws_size,
                              hipStream_t stream) {
  (void)in_sizes; (void)n_in; (void)out_size; (void)ws_size;

  const float* x     = (const float*)d_in[0];
  // d_in[1] = loop_cnt (unused)
  const float* W_ih0 = (const float*)d_in[2];
  const float* W_hh0 = (const float*)d_in[3];
  const float* b_ih0 = (const float*)d_in[4];
  const float* b_hh0 = (const float*)d_in[5];
  const float* W_ih1 = (const float*)d_in[6];
  const float* W_hh1 = (const float*)d_in[7];
  const float* b_ih1 = (const float*)d_in[8];
  const float* b_hh1 = (const float*)d_in[9];
  const float* W_fc  = (const float*)d_in[10];
  const float* b_fc  = (const float*)d_in[11];

  char* ws = (char*)d_ws;
  __bf16* hs0   = (__bf16*)(ws + OFF_HS0);
  __bf16* wih0b = (__bf16*)(ws + OFF_WIH0);
  __bf16* whh0b = (__bf16*)(ws + OFF_WHH0);
  __bf16* wih1b = (__bf16*)(ws + OFF_WIH1);
  __bf16* whh1b = (__bf16*)(ws + OFF_WHH1);
  float*  bias0 = (float*)(ws + OFF_B0);
  float*  bias1 = (float*)(ws + OFF_B1);

  auto blocks = [](int n) { return (n + 255) / 256; };

  // Convert weights to bf16 once per call; combine biases.
  cvt_f32_bf16<<<blocks(4 * HH * II), 256, 0, stream>>>(W_ih0, wih0b, 4 * HH * II);
  cvt_f32_bf16<<<blocks(4 * HH * HH), 256, 0, stream>>>(W_hh0, whh0b, 4 * HH * HH);
  cvt_f32_bf16<<<blocks(4 * HH * HH), 256, 0, stream>>>(W_ih1, wih1b, 4 * HH * HH);
  cvt_f32_bf16<<<blocks(4 * HH * HH), 256, 0, stream>>>(W_hh1, whh1b, 4 * HH * HH);
  bias_combine<<<blocks(4 * HH), 256, 0, stream>>>(b_ih0, b_hh0, bias0, 4 * HH);
  bias_combine<<<blocks(4 * HH), 256, 0, stream>>>(b_ih1, b_hh1, bias1, 4 * HH);

  // Layer 0: f32 input x -> bf16 hidden sequence hs0.
  lstm_layer<II, true, true, false><<<BB / 16, 512, 0, stream>>>(
      x, nullptr, wih0b, whh0b, bias0, hs0, nullptr, nullptr, nullptr);

  // Layer 1: hs0 -> final hidden state, fused FC into d_out [B,1].
  lstm_layer<HH, false, false, true><<<BB / 16, 512, 0, stream>>>(
      nullptr, hs0, wih1b, whh1b, bias1, nullptr, W_fc, b_fc, (float*)d_out);
}
